// Res_Slim_ViT_Adaptive_15298673508441
// MI455X (gfx1250) — matrix operand order repaired and checked by reading, not compiled, not run on hardware
//
#include <hip/hip_runtime.h>
#include <math.h>
#include <stdint.h>

// ---------------------------------------------------------------------------
// Problem constants (match reference)
// ---------------------------------------------------------------------------
#define B_    2
#define V_    4
#define H_    256
#define W_    512
#define P_    16
#define D_    1024
#define NH_   16
#define HD_   64
#define L_    512      // (H/P)*(W/P) = 16*32
#define KEEP_ 128
#define MAG_  4
#define DEPTH_ 6

typedef __attribute__((ext_vector_type(16))) __bf16 bf16x16;
typedef __attribute__((ext_vector_type(8)))  __bf16 bf16x8;
typedef __attribute__((ext_vector_type(8)))  float  f32x8;
typedef __attribute__((ext_vector_type(4)))  unsigned int u32x4;
typedef __attribute__((ext_vector_type(8)))  int          i32x8;
typedef __attribute__((ext_vector_type(4)))  int          i32x4;

#if defined(__AMDGCN__) && __has_builtin(__builtin_amdgcn_tensor_load_to_lds)
#define USE_TDM 1
#endif

__device__ __forceinline__ __bf16 f2bf(float x) {
  unsigned u = __builtin_bit_cast(unsigned, x);
  u += 0x7FFFu + ((u >> 16) & 1u);               // round-to-nearest-even
  unsigned short hs = (unsigned short)(u >> 16);
  return __builtin_bit_cast(__bf16, hs);
}
__device__ __forceinline__ float gelu_f(float x) {
  return 0.5f * x * (1.0f + erff(x * 0.70710678118654752f));
}

#ifdef USE_TDM
// Issue a TDM load of a 64-row x 32-element bf16 tile (row stride K elements)
// into LDS, inserting 16B of padding after each 64B row (=> [64][40] layout).
__device__ __forceinline__ void tdm_issue_b_tile(const __bf16* lds_dst,
                                                 const __bf16* gsrc, int K)
{
  // Generic LDS pointer: low 32 bits are the LDS byte offset (aperture rule).
  unsigned lds_off = (unsigned)(unsigned long long)(uintptr_t)lds_dst;
  unsigned long long g = (unsigned long long)(uintptr_t)gsrc;
  u32x4 g0;
  g0.x = 1u;                                        // count=1, user mode
  g0.y = lds_off;                                   // lds_addr
  g0.z = (unsigned)g;                               // global_addr[31:0]
  g0.w = (unsigned)((g >> 32) & 0x01FFFFFFu) | (2u << 30);  // addr[56:32], type=2
  i32x8 g1;
  // data_size=2B, pad_enable, pad_interval=16 DWORDs, pad_amount=4 DWORDs
  g1[0] = (int)((1u << 16) | (1u << 20) | (3u << 22) | (3u << 25));
  g1[1] = (int)(((unsigned)K & 0xFFFFu) << 16);     // tensor_dim0[15:0]
  g1[2] = (int)(((unsigned)K >> 16));               // dim0 hi | dim1 lo (dim1=1<<20)
  g1[3] = (int)(16u | (32u << 16));                 // dim1 hi=16, tile_dim0=32
  g1[4] = 64;                                       // tile_dim1=64, tile_dim2=0
  g1[5] = K;                                        // tensor_dim0_stride lo
  g1[6] = 0;
  g1[7] = 0;
  i32x4 z4 = {0, 0, 0, 0};
  i32x8 z8 = {0, 0, 0, 0, 0, 0, 0, 0};
  // 6-arg variant (clang-23 / therock-10.0 headers)
  __builtin_amdgcn_tensor_load_to_lds(g0, g1, z4, z4, z8, 0);
}
#endif

// ---------------------------------------------------------------------------
// WMMA GEMM:  C[M,N] = act(A[M,K] @ B[K,N] + bias) + res
//   - A,res,C row-major f32; B passed PRE-TRANSPOSED+bf16 as Bt[N][K]
//   - requires M%128==0, N%64==0, K%32==0 (true for every GEMM in this net)
//   - act: 0 = none, 1 = exact GELU;  bias/res may be nullptr
// Block: 128 threads = 4 waves; tile 128x64x32; each wave: 32 rows x 64 cols,
// 8 accumulators -> 8 v_wmma per K-step. B tiles double-buffered via TDM.
// ---------------------------------------------------------------------------
__global__ __launch_bounds__(128)
void gemm_bf16_wmma(const float* __restrict__ A, const __bf16* __restrict__ Bt,
                    const float* __restrict__ bias, const float* __restrict__ res,
                    float* __restrict__ C, int M, int N, int K, int act)
{
  __shared__ __bf16 lA[128][40];     // 128 rows x 32 K (+8 pad, 16B aligned)
  __shared__ __bf16 lB[2][64][40];   // double-buffered Bt tile [n][k]

  const int bm  = blockIdx.y * 128;
  const int bn  = blockIdx.x * 64;
  const int tid = threadIdx.x;
  const int wv   = tid >> 5;
  const int lane = tid & 31;
  const int hf   = lane >> 4;      // lane half (K-half selector)
  const int mn   = lane & 15;      // row (A) / col (B) within 16

  f32x8 acc[2][4] = {};
  const int nsteps = K >> 5;

#ifdef USE_TDM
  if (wv == 0)
    tdm_issue_b_tile(&lB[0][0][0], Bt + (size_t)bn * K, K);
#endif

  for (int s = 0; s < nsteps; ++s) {
    const int k0  = s << 5;
    const int par = s & 1;

    // stage A tile (128x32) as bf16, coalesced along K
    for (int i = tid; i < 4096; i += 128) {
      int r = i >> 5, c = i & 31;
      lA[r][c] = f2bf(A[(size_t)(bm + r) * K + (k0 + c)]);
    }
#ifdef USE_TDM
    if (wv == 0) __builtin_amdgcn_s_wait_tensorcnt(0);
#else
    for (int i = tid; i < 2048; i += 128) {
      int n = i >> 5, c = i & 31;
      lB[par][n][c] = Bt[(size_t)(bn + n) * K + (k0 + c)];
    }
#endif
    __syncthreads();   // A + current B tile visible to all waves

#ifdef USE_TDM
    if (wv == 0 && s + 1 < nsteps)
      tdm_issue_b_tile(&lB[par ^ 1][0][0], Bt + (size_t)bn * K + (k0 + 32), K);
#endif

    // A fragments: two 16-row groups per wave
    bf16x16 af[2];
#pragma unroll
    for (int g = 0; g < 2; ++g) {
      const __bf16* p = &lA[wv * 32 + g * 16 + mn][hf * 8];
      bf16x8 lo = *(const bf16x8*)p;
      bf16x8 hi = *(const bf16x8*)(p + 16);
      af[g] = __builtin_shufflevector(lo, hi, 0,1,2,3,4,5,6,7,8,9,10,11,12,13,14,15);
    }
    // B fragments: lane half hf covers K = hf*8 .. hf*8+7 and 16+hf*8 .. 16+hf*8+7 (same as A)
#pragma unroll
    for (int t = 0; t < 4; ++t) {
      const __bf16* p = &lB[par][t * 16 + mn][hf * 8];
      bf16x8 lo = *(const bf16x8*)p;
      bf16x8 hi = *(const bf16x8*)(p + 16);
      bf16x16 bfv = __builtin_shufflevector(lo, hi, 0,1,2,3,4,5,6,7,8,9,10,11,12,13,14,15);
      acc[0][t] = __builtin_amdgcn_wmma_f32_16x16x32_bf16(
          false, af[0], false, bfv, (short)0, acc[0][t], false, false);
      acc[1][t] = __builtin_amdgcn_wmma_f32_16x16x32_bf16(
          false, af[1], false, bfv, (short)0, acc[1][t], false, false);
    }
    __syncthreads();   // protect lA / lB[par] before next overwrite
  }

  // Epilogue: D layout -> VGPR r holds M = r + 8*half, lane holds N = mn
#pragma unroll
  for (int g = 0; g < 2; ++g)
#pragma unroll
    for (int t = 0; t < 4; ++t) {
      int col = bn + t * 16 + mn;
      float bv = bias ? bias[col] : 0.0f;
#pragma unroll
      for (int r = 0; r < 8; ++r) {
        int row = bm + wv * 32 + g * 16 + hf * 8 + r;
        float val = acc[g][t][r] + bv;
        if (act) val = gelu_f(val);
        if (res) val += res[(size_t)row * N + col];
        C[(size_t)row * N + col] = val;
      }
    }
}

// ---------------------------------------------------------------------------
// Tiled transpose + f32->bf16 convert:  Wt[n][k] = bf16(W[k][n])
// ---------------------------------------------------------------------------
__global__ __launch_bounds__(256)
void wtrans_kernel(const float* __restrict__ W, __bf16* __restrict__ Wt,
                   int K, int N)
{
  __shared__ float tile[32][33];
  const int bx = blockIdx.x * 32;   // n base
  const int by = blockIdx.y * 32;   // k base
  const int tx = threadIdx.x & 31, ty = threadIdx.x >> 5;
  for (int r = ty; r < 32; r += 8)
    tile[r][tx] = W[(size_t)(by + r) * N + (bx + tx)];
  __syncthreads();
  for (int r = ty; r < 32; r += 8)
    Wt[(size_t)(bx + r) * K + (by + tx)] = f2bf(tile[tx][r]);
}

// ---------------------------------------------------------------------------
// LayerNorm over D=1024 per row; one block per row.
// ---------------------------------------------------------------------------
__global__ __launch_bounds__(256)
void ln_kernel(const float* __restrict__ x, const float* __restrict__ g,
               const float* __restrict__ b, float* __restrict__ y)
{
  const int r = blockIdx.x, t = threadIdx.x;
  __shared__ float red[256];
  const float* xr = x + (size_t)r * D_;
  float v[4], s = 0.f;
#pragma unroll
  for (int i = 0; i < 4; ++i) { v[i] = xr[t + 256 * i]; s += v[i]; }
  red[t] = s; __syncthreads();
  for (int k = 128; k > 0; k >>= 1) { if (t < k) red[t] += red[t + k]; __syncthreads(); }
  float mean = red[0] * (1.0f / D_); __syncthreads();
  float s2 = 0.f;
#pragma unroll
  for (int i = 0; i < 4; ++i) { float d = v[i] - mean; s2 += d * d; }
  red[t] = s2; __syncthreads();
  for (int k = 128; k > 0; k >>= 1) { if (t < k) red[t] += red[t + k]; __syncthreads(); }
  float rstd = rsqrtf(red[0] * (1.0f / D_) + 1e-5f);
  float* yr = y + (size_t)r * D_;
#pragma unroll
  for (int i = 0; i < 4; ++i) { int c = t + 256 * i; yr[c] = (v[i] - mean) * rstd * g[c] + b[c]; }
}

// ---------------------------------------------------------------------------
// Self-attention from packed qkv (rows = b*n+t, cols = (c*NH+h)*HD+d)
// One block (64 threads) per (query token, head, batch).
// ---------------------------------------------------------------------------
__global__ __launch_bounds__(64)
void attn_kernel(const float* __restrict__ qkv, float* __restrict__ o, int n)
{
  const int q = blockIdx.x, h = blockIdx.y, b = blockIdx.z, t = threadIdx.x;
  __shared__ float qs[HD_];
  __shared__ float sc[L_];      // n <= 512
  __shared__ float red[64];
  const size_t ld = 3 * D_;
  qs[t] = qkv[((size_t)(b * n + q)) * ld + h * HD_ + t];
  __syncthreads();
  float mx = -1e30f;
  for (int k = t; k < n; k += 64) {
    const float* krow = qkv + ((size_t)(b * n + k)) * ld + (NH_ + h) * HD_;
    float s = 0.f;
#pragma unroll 16
    for (int d = 0; d < HD_; ++d) s += qs[d] * krow[d];
    s *= 0.125f;                 // HD^-0.5
    sc[k] = s;
    mx = fmaxf(mx, s);
  }
  red[t] = mx; __syncthreads();
  for (int k = 32; k > 0; k >>= 1) { if (t < k) red[t] = fmaxf(red[t], red[t + k]); __syncthreads(); }
  mx = red[0]; __syncthreads();
  float sum = 0.f;
  for (int k = t; k < n; k += 64) { float e = __expf(sc[k] - mx); sc[k] = e; sum += e; }
  red[t] = sum; __syncthreads();
  for (int k = 32; k > 0; k >>= 1) { if (t < k) red[t] += red[t + k]; __syncthreads(); }
  float inv = 1.0f / red[0]; __syncthreads();
  float acc = 0.f;
  for (int k = 0; k < n; ++k)
    acc += sc[k] * qkv[((size_t)(b * n + k)) * ld + (2 * NH_ + h) * HD_ + t];
  o[((size_t)(b * n + q)) * D_ + h * HD_ + t] = acc * inv;
}

// ---------------------------------------------------------------------------
// Variable-aggregation cross attention (V=4 keys per token, per head)
// ---------------------------------------------------------------------------
__global__ __launch_bounds__(64)
void agg_attn_kernel(const float* __restrict__ qv, const float* __restrict__ kb,
                     const float* __restrict__ vb, float* __restrict__ out)
{
  const int nidx = blockIdx.x, h = blockIdx.y, t = threadIdx.x;
  __shared__ float red[64];
  __shared__ float p[V_];
  const float q = qv[h * HD_ + t];
  for (int v = 0; v < V_; ++v) {
    red[t] = q * kb[((size_t)nidx * V_ + v) * D_ + h * HD_ + t];
    __syncthreads();
    for (int k = 32; k > 0; k >>= 1) { if (t < k) red[t] += red[t + k]; __syncthreads(); }
    if (t == 0) p[v] = red[0] * 0.125f;
    __syncthreads();
  }
  if (t == 0) {
    float mx = fmaxf(fmaxf(p[0], p[1]), fmaxf(p[2], p[3]));
    float s = 0.f;
    for (int v = 0; v < V_; ++v) { p[v] = __expf(p[v] - mx); s += p[v]; }
    for (int v = 0; v < V_; ++v) p[v] /= s;
  }
  __syncthreads();
  float acc = 0.f;
  for (int v = 0; v < V_; ++v)
    acc += p[v] * vb[((size_t)nidx * V_ + v) * D_ + h * HD_ + t];
  out[(size_t)nidx * D_ + h * HD_ + t] = acc;
}

// ---------------------------------------------------------------------------
// Small elementwise / reshaping kernels
// ---------------------------------------------------------------------------
__global__ __launch_bounds__(256)
void im2patch_kernel(const float* __restrict__ x, float* __restrict__ xp2)
{ // xp2 layout (V, B*L, 256)
  int i = blockIdx.x * 256 + threadIdx.x;
  if (i >= V_ * B_ * L_ * 256) return;
  int e  = i & 255;
  int bl = (i >> 8) & (B_ * L_ - 1);
  int v  = i >> 18;
  int b  = bl >> 9;
  int l  = bl & 511;
  int hh = l >> 5, wwi = l & 31;
  int pi = e >> 4, qi  = e & 15;
  xp2[i] = x[(((size_t)(b * V_ + v) * H_) + hh * P_ + pi) * W_ + wwi * P_ + qi];
}

__global__ void combine_bias_kernel(const float* a, const float* b, float* o)
{ int i = blockIdx.x * 256 + threadIdx.x; if (i < V_ * D_) o[i] = a[i] + b[i]; }

__global__ __launch_bounds__(256)
void transpose_vf_kernel(const float* __restrict__ tok2, float* __restrict__ xf)
{ // tok2 (V, B*L, D) -> xf ((B*L)*V, D)
  size_t i = (size_t)blockIdx.x * 256 + threadIdx.x;
  if (i >= (size_t)B_ * L_ * V_ * D_) return;
  int d = (int)(i & (D_ - 1));
  size_t r = i >> 10;
  int v = (int)(r & 3);
  size_t bl = r >> 2;
  xf[i] = tok2[((size_t)v * (B_ * L_) + bl) * D_ + d];
}

__global__ void vecmat_kernel(const float* __restrict__ vec, const float* __restrict__ Wm,
                              float* __restrict__ out, int K, int N)
{
  int n = blockIdx.x * 256 + threadIdx.x;
  if (n >= N) return;
  float s = 0.f;
  for (int k = 0; k < K; ++k) s += vec[k] * Wm[(size_t)k * N + n];
  out[n] = s;
}

__global__ void add_pos_kernel(const float* a, const float* pos, float* o)
{
  size_t i = (size_t)blockIdx.x * 256 + threadIdx.x;
  if (i >= (size_t)B_ * L_ * D_) return;
  int d = (int)(i & (D_ - 1));
  int l = (int)((i >> 10) & (L_ - 1));
  o[i] = a[i] + pos[(size_t)l * D_ + d];
}

__global__ __launch_bounds__(256)
void gather_kernel(const float* __restrict__ xt, const int* __restrict__ idx,
                   float* __restrict__ hard)
{
  int r = blockIdx.x;                 // 0 .. B*KEEP-1
  int b = r >> 7, j = r & (KEEP_ - 1);
  int src = idx[b * KEEP_ + j];
  const float* s = xt + ((size_t)b * L_ + src) * D_;
  float* d = hard + (size_t)r * D_;
  for (int i = threadIdx.x; i < D_; i += 256) d[i] = s[i];
}

__global__ __launch_bounds__(256)
void scatter_kernel(const float* __restrict__ hard, const int* __restrict__ idx,
                    float* __restrict__ xt)
{
  int r = blockIdx.x;
  int b = r >> 7, j = r & (KEEP_ - 1);
  int dst = idx[b * KEEP_ + j];
  const float* s = hard + (size_t)r * D_;
  float* d = xt + ((size_t)b * L_ + dst) * D_;
  for (int i = threadIdx.x; i < D_; i += 256) d[i] = s[i];
}

__global__ __launch_bounds__(256)
void unpatchify_kernel(const float* __restrict__ pred, float* __restrict__ y)
{ // pred (B, L, 4096) -> y (B, 1024, 2048)
  size_t i = (size_t)blockIdx.x * 256 + threadIdx.x;
  if (i >= (size_t)B_ * 1024 * 2048) return;
  int col = (int)(i & 2047);
  int row = (int)((i >> 11) & 1023);
  int b   = (int)(i >> 21);
  int h2i = row >> 4, pi = row & 15;
  int w2i = col >> 4, qi = col & 15;
  int l = h2i * 8 + (w2i >> 4);
  int e = (w2i & 15) * 256 + pi * 16 + qi;
  y[i] = pred[((size_t)b * L_ + l) * 4096 + e];
}

// ---------------------------------------------------------------------------
// Conv path (memory bound, scalar)
// ---------------------------------------------------------------------------
__global__ __launch_bounds__(256)
void conv1_kernel(const float* __restrict__ x, const float* __restrict__ w,
                  const float* __restrict__ bias, float* __restrict__ out)
{ // (B,4,256,512) -> gelu(conv3x3) -> (B,64,256,512)
  size_t i = (size_t)blockIdx.x * 256 + threadIdx.x;
  if (i >= (size_t)B_ * 64 * H_ * W_) return;
  int xx = (int)(i % W_);
  int yy = (int)((i / W_) % H_);
  int oc = (int)((i / ((size_t)W_ * H_)) % 64);
  int b  = (int)(i / ((size_t)W_ * H_ * 64));
  float acc = bias[oc];
  for (int ic = 0; ic < V_; ++ic)
    for (int dy = 0; dy < 3; ++dy) {
      int sy = yy + dy - 1; if (sy < 0 || sy >= H_) continue;
      for (int dx = 0; dx < 3; ++dx) {
        int sx = xx + dx - 1; if (sx < 0 || sx >= W_) continue;
        acc += w[((oc * V_ + ic) * 3 + dy) * 3 + dx] *
               x[(((size_t)b * V_ + ic) * H_ + sy) * W_ + sx];
      }
    }
  out[i] = gelu_f(acc);
}

__global__ __launch_bounds__(256)
void conv2_fused_kernel(const float* __restrict__ c1, const float* __restrict__ w,
                        const float* __restrict__ b2, const float* __restrict__ y,
                        float* __restrict__ s)
{ // pixel-shuffle(r=4) of c1 indexed on the fly; conv3x3 (4->1); + y
  size_t i = (size_t)blockIdx.x * 256 + threadIdx.x;
  if (i >= (size_t)B_ * 1024 * 2048) return;
  int xx = (int)(i & 2047);
  int yy = (int)((i >> 11) & 1023);
  int b  = (int)(i >> 21);
  float acc = b2[0];
  for (int c2 = 0; c2 < 4; ++c2)
    for (int dy = 0; dy < 3; ++dy) {
      int sy = yy + dy - 1; if (sy < 0 || sy >= 1024) continue;
      for (int dx = 0; dx < 3; ++dx) {
        int sx = xx + dx - 1; if (sx < 0 || sx >= 2048) continue;
        int c = c2 * 16 + (sy & 3) * 4 + (sx & 3);
        acc += w[(c2 * 3 + dy) * 3 + dx] *
               c1[(((size_t)b * 64 + c) * H_ + (sy >> 2)) * W_ + (sx >> 2)];
      }
    }
  s[i] = acc + y[i];
}

__global__ __launch_bounds__(256)
void conv3_kernel(const float* __restrict__ s, const float* __restrict__ w,
                  const float* __restrict__ cb, float* __restrict__ out)
{
  size_t i = (size_t)blockIdx.x * 256 + threadIdx.x;
  if (i >= (size_t)B_ * 1024 * 2048) return;
  int xx = (int)(i & 2047);
  int yy = (int)((i >> 11) & 1023);
  int b  = (int)(i >> 21);
  float acc = cb[0];
  for (int dy = 0; dy < 3; ++dy) {
    int sy = yy + dy - 1; if (sy < 0 || sy >= 1024) continue;
    for (int dx = 0; dx < 3; ++dx) {
      int sx = xx + dx - 1; if (sx < 0 || sx >= 2048) continue;
      acc += w[dy * 3 + dx] * s[((size_t)b * 1024 + sy) * 2048 + sx];
    }
  }
  out[i] = acc;
}

// ---------------------------------------------------------------------------
// Host-side orchestration
// ---------------------------------------------------------------------------
static inline unsigned eg(size_t n) { return (unsigned)((n + 255) / 256); }

static void run_block(const float* l1w, const float* l1b,
                      const __bf16* qwt, const float* qb,
                      const __bf16* pwt, const float* pb,
                      const float* l2w, const float* l2b,
                      const __bf16* f1wt, const float* f1b,
                      const __bf16* f2wt, const float* f2b,
                      float* xt, float* hn, float* qkvb, float* ob, float* mlp,
                      int bsz, int n, hipStream_t st)
{
  const int M = bsz * n;
  ln_kernel<<<M, 256, 0, st>>>(xt, l1w, l1b, hn);
  gemm_bf16_wmma<<<dim3(3 * D_ / 64, M / 128), 128, 0, st>>>(
      hn, qwt, qb, nullptr, qkvb, M, 3 * D_, D_, 0);
  attn_kernel<<<dim3(n, NH_, bsz), 64, 0, st>>>(qkvb, ob, n);
  gemm_bf16_wmma<<<dim3(D_ / 64, M / 128), 128, 0, st>>>(
      ob, pwt, pb, xt, xt, M, D_, D_, 0);
  ln_kernel<<<M, 256, 0, st>>>(xt, l2w, l2b, hn);
  gemm_bf16_wmma<<<dim3(4 * D_ / 64, M / 128), 128, 0, st>>>(
      hn, f1wt, f1b, nullptr, mlp, M, 4 * D_, D_, 1);
  gemm_bf16_wmma<<<dim3(D_ / 64, M / 128), 128, 0, st>>>(
      mlp, f2wt, f2b, xt, xt, M, D_, 4 * D_, 0);
}

extern "C" void kernel_launch(void* const* d_in, const int* in_sizes, int n_in,
                              void* d_out, int out_size, void* d_ws, size_t ws_size,
                              hipStream_t stream)
{
  (void)in_sizes; (void)n_in; (void)out_size; (void)ws_size;
  const float* x          = (const float*)d_in[0];
  const int*   hard_idx   = (const int*)  d_in[1];
  const float* patch_w    = (const float*)d_in[2];
  const float* patch_b    = (const float*)d_in[3];
  const float* var_embed  = (const float*)d_in[4];
  const float* var_query  = (const float*)d_in[5];
  const float* agg_wq     = (const float*)d_in[6];
  const float* agg_wk     = (const float*)d_in[7];
  const float* agg_wv     = (const float*)d_in[8];
  const float* agg_wo     = (const float*)d_in[9];
  const float* agg_bo     = (const float*)d_in[10];
  const float* pos_embed  = (const float*)d_in[11];
  const float* ln1_w      = (const float*)d_in[12];
  const float* ln1_b      = (const float*)d_in[13];
  const float* qkv_w      = (const float*)d_in[14];
  const float* qkv_b      = (const float*)d_in[15];
  const float* aproj_w    = (const float*)d_in[16];
  const float* aproj_b    = (const float*)d_in[17];
  const float* ln2_w      = (const float*)d_in[18];
  const float* ln2_b      = (const float*)d_in[19];
  const float* fc1_w      = (const float*)d_in[20];
  const float* fc1_b      = (const float*)d_in[21];
  const float* fc2_w      = (const float*)d_in[22];
  const float* fc2_b      = (const float*)d_in[23];
  const float* norm_w     = (const float*)d_in[24];
  const float* norm_b     = (const float*)d_in[25];
  const float* head_w     = (const float*)d_in[26];
  const float* head_b     = (const float*)d_in[27];
  const float* head_out_w = (const float*)d_in[28];
  const float* head_out_b = (const float*)d_in[29];
  const float* p2c1_w     = (const float*)d_in[30];
  const float* p2c1_b     = (const float*)d_in[31];
  const float* p2c2_w     = (const float*)d_in[32];
  const float* p2c2_b     = (const float*)d_in[33];
  const float* cout_w     = (const float*)d_in[34];
  const float* cout_b     = (const float*)d_in[35];
  float* out = (float*)d_out;

  // ---- workspace layout ----
  float* ws = (float*)d_ws;
  size_t off = 0;
  auto alloc = [&](size_t n) { float* p = ws + off; off += n; return p; };
  const size_t BL = (size_t)B_ * L_;
  float* xp2   = alloc((size_t)V_ * BL * 256);   // (V, B*L, 256)
  float* tok2  = alloc((size_t)V_ * BL * D_);    // (V, B*L, D)
  float* pb2   = alloc((size_t)V_ * D_);
  float* xf    = alloc(BL * V_ * D_);            // ((B*L)*V, D)
  float* kb    = alloc(BL * V_ * D_);
  float* vb    = alloc(BL * V_ * D_);
  float* qvec  = alloc(D_);
  float* aggb  = alloc(BL * D_);
  float* xt    = alloc(BL * D_);
  float* hn    = alloc(BL * D_);
  float* qkvb  = alloc(BL * 3 * D_);
  float* ob    = alloc(BL * D_);
  float* mlp   = alloc(BL * 4 * D_);             // also reused as pred
  float* hard  = alloc((size_t)B_ * KEEP_ * D_);
  float* ybuf  = alloc((size_t)B_ * 1024 * 2048);
  float* c1    = alloc((size_t)B_ * 64 * H_ * W_);
  float* pred  = mlp;                             // (B*L, 4096)
  float* sbuf  = xf;                              // reuse: xf dead after k/v GEMMs

  off = (off + 3) & ~(size_t)3;                   // 16B align for bf16 region
  auto allocbf = [&](size_t n) { __bf16* p = (__bf16*)(ws + off); off += n / 2; return p; };
  __bf16* patch_wt    = allocbf((size_t)V_ * D_ * 256);       // per v: [1024][256]
  __bf16* agg_wkt     = allocbf((size_t)D_ * D_);
  __bf16* agg_wvt     = allocbf((size_t)D_ * D_);
  __bf16* agg_wot     = allocbf((size_t)D_ * D_);
  __bf16* qkv_wt      = allocbf((size_t)DEPTH_ * 3 * D_ * D_);
  __bf16* aproj_wt    = allocbf((size_t)DEPTH_ * D_ * D_);
  __bf16* fc1_wt      = allocbf((size_t)DEPTH_ * 4 * D_ * D_);
  __bf16* fc2_wt      = allocbf((size_t)DEPTH_ * 4 * D_ * D_);
  __bf16* head_wt     = allocbf((size_t)2 * D_ * D_);
  __bf16* head_out_wt = allocbf((size_t)4096 * D_);

  // ---- 0. pre-transpose + bf16-convert all GEMM weights (once per launch) ----
  auto trans = [&](const float* W, __bf16* Wt, int K, int N) {
    wtrans_kernel<<<dim3(N / 32, K / 32), 256, 0, stream>>>(W, Wt, K, N);
  };
  for (int v = 0; v < V_; ++v)
    trans(patch_w + (size_t)v * 256 * D_, patch_wt + (size_t)v * D_ * 256, 256, D_);
  trans(agg_wk, agg_wkt, D_, D_);
  trans(agg_wv, agg_wvt, D_, D_);
  trans(agg_wo, agg_wot, D_, D_);
  for (int i = 0; i < DEPTH_; ++i) {
    trans(qkv_w + (size_t)i * D_ * 3 * D_, qkv_wt + (size_t)i * 3 * D_ * D_, D_, 3 * D_);
    trans(aproj_w + (size_t)i * D_ * D_, aproj_wt + (size_t)i * D_ * D_, D_, D_);
    trans(fc1_w + (size_t)i * D_ * 4 * D_, fc1_wt + (size_t)i * 4 * D_ * D_, D_, 4 * D_);
    trans(fc2_w + (size_t)i * 4 * D_ * D_, fc2_wt + (size_t)i * 4 * D_ * D_, 4 * D_, D_);
  }
  trans(head_w, head_wt, D_, D_);
  trans(head_w + (size_t)D_ * D_, head_wt + (size_t)D_ * D_, D_, D_);
  trans(head_out_w, head_out_wt, D_, 4096);

  // ---- 1. patch embedding (per variable GEMM) ----
  im2patch_kernel<<<eg((size_t)V_ * BL * 256), 256, 0, stream>>>(x, xp2);
  combine_bias_kernel<<<eg(V_ * D_), 256, 0, stream>>>(patch_b, var_embed, pb2);
  for (int v = 0; v < V_; ++v)
    gemm_bf16_wmma<<<dim3(D_ / 64, (int)BL / 128), 128, 0, stream>>>(
        xp2 + (size_t)v * BL * 256, patch_wt + (size_t)v * D_ * 256,
        pb2 + v * D_, nullptr, tok2 + (size_t)v * BL * D_,
        (int)BL, D_, 256, 0);

  // ---- 2. aggregate variables (cross-attention over V) ----
  transpose_vf_kernel<<<eg(BL * V_ * D_), 256, 0, stream>>>(tok2, xf);
  gemm_bf16_wmma<<<dim3(D_ / 64, (int)(BL * V_) / 128), 128, 0, stream>>>(
      xf, agg_wkt, nullptr, nullptr, kb, (int)(BL * V_), D_, D_, 0);
  gemm_bf16_wmma<<<dim3(D_ / 64, (int)(BL * V_) / 128), 128, 0, stream>>>(
      xf, agg_wvt, nullptr, nullptr, vb, (int)(BL * V_), D_, D_, 0);
  vecmat_kernel<<<eg(D_), 256, 0, stream>>>(var_query, agg_wq, qvec, D_, D_);
  agg_attn_kernel<<<dim3((unsigned)BL, NH_), 64, 0, stream>>>(qvec, kb, vb, aggb);
  gemm_bf16_wmma<<<dim3(D_ / 64, (int)BL / 128), 128, 0, stream>>>(
      aggb, agg_wot, agg_bo, nullptr, ob, (int)BL, D_, D_, 0);
  add_pos_kernel<<<eg(BL * D_), 256, 0, stream>>>(ob, pos_embed, xt);

  // ---- 3. transformer blocks ----
  auto blk = [&](int i, float* buf, int n) {
    run_block(ln1_w + i * D_, ln1_b + i * D_,
              qkv_wt + (size_t)i * 3 * D_ * D_, qkv_b + (size_t)i * 3 * D_,
              aproj_wt + (size_t)i * D_ * D_, aproj_b + i * D_,
              ln2_w + i * D_, ln2_b + i * D_,
              fc1_wt + (size_t)i * 4 * D_ * D_, fc1_b + (size_t)i * 4 * D_,
              fc2_wt + (size_t)i * 4 * D_ * D_, fc2_b + i * D_,
              buf, hn, qkvb, ob, mlp, B_, n, stream);
  };
  blk(0, xt, L_);
  blk(1, xt, L_);
  gather_kernel<<<B_ * KEEP_, 256, 0, stream>>>(xt, hard_idx, hard);
  blk(2, hard, KEEP_);
  blk(3, hard, KEEP_);
  blk(4, hard, KEEP_);
  scatter_kernel<<<B_ * KEEP_, 256, 0, stream>>>(hard, hard_idx, xt);
  blk(5, xt, L_);

  // ---- 4. head ----
  ln_kernel<<<(int)BL, 256, 0, stream>>>(xt, norm_w, norm_b, hn);
  gemm_bf16_wmma<<<dim3(D_ / 64, (int)BL / 128), 128, 0, stream>>>(
      hn, head_wt, head_b, nullptr, ob, (int)BL, D_, D_, 1);
  gemm_bf16_wmma<<<dim3(D_ / 64, (int)BL / 128), 128, 0, stream>>>(
      ob, head_wt + (size_t)D_ * D_, head_b + D_, nullptr, hn, (int)BL, D_, D_, 1);
  gemm_bf16_wmma<<<dim3(4096 / 64, (int)BL / 128), 128, 0, stream>>>(
      hn, head_out_wt, head_out_b, nullptr, pred, (int)BL, 4096, D_, 0);

  // ---- 5. unpatchify + conv residual path ----
  unpatchify_kernel<<<eg((size_t)B_ * 1024 * 2048), 256, 0, stream>>>(pred, ybuf);
  conv1_kernel<<<eg((size_t)B_ * 64 * H_ * W_), 256, 0, stream>>>(x, p2c1_w, p2c1_b, c1);
  conv2_fused_kernel<<<eg((size_t)B_ * 1024 * 2048), 256, 0, stream>>>(
      c1, p2c2_w, p2c2_b, ybuf, sbuf);
  conv3_kernel<<<eg((size_t)B_ * 1024 * 2048), 256, 0, stream>>>(sbuf, cout_w, cout_b, out);
}
